// GRUModule_79980880986579
// MI455X (gfx1250) — matrix operand order repaired and checked by reading, not compiled, not run on hardware
//
#include <hip/hip_runtime.h>
#include <hip/hip_bf16.h>
#include <cstdint>

typedef __bf16 bf16;
typedef bf16  v16bf __attribute__((ext_vector_type(16)));
typedef float v8f   __attribute__((ext_vector_type(8)));
typedef unsigned int u32x4 __attribute__((ext_vector_type(4)));
typedef float f32x4 __attribute__((ext_vector_type(4)));

#define B_   128
#define T_   1024
#define IN_  64
#define H_   128
#define G3_  384
#define DIM_ 64

union FragU { v16bf v; u32x4 q[2]; };
union Pack8 { bf16 h[8]; u32x4 q; };

// A fragment (16x32 bf16, MxK) from row-major [.. x ld] buffer at (m0=0, k0)=p.
// ISA layout: lane L holds row M=L%16; VGPR0-3: K=(L/16)*8..+8, VGPR4-7: K=16+(L/16)*8..+8
static __device__ inline v16bf frag_a_load(const bf16* p, int ld, int lane) {
  int m = lane & 15, kh = lane >> 4;
  const bf16* r = p + (size_t)m * ld + kh * 8;
  FragU f;
  f.q[0] = *(const u32x4*)(r);
  f.q[1] = *(const u32x4*)(r + 16);
  return f.v;
}
// B fragment (32x16 bf16, KxN) where memory holds W[N][K] row-major (so B(k,n)=W[n0+n][k0+k]).
// ISA layout: lane L holds column N=L%16; VGPR0-3: K=(L/16)*8..+8, VGPR4-7: K=16+(L/16)*8..+8
static __device__ inline v16bf frag_b_load(const bf16* p, int ld, int lane) {
  int n = lane & 15, kh = lane >> 4;
  const bf16* r = p + (size_t)n * ld + kh * 8;
  FragU f;
  f.q[0] = *(const u32x4*)(r);
  f.q[1] = *(const u32x4*)(r + 16);
  return f.v;
}
static __device__ inline v8f wmma_bf16(v16bf a, v16bf b, v8f c) {
  // D = A x B + C, f32 accumulate
  return __builtin_amdgcn_wmma_f32_16x16x32_bf16(false, a, false, b, (short)0, c, false, false);
}
static __device__ inline v8f v8f_zero() {
  v8f c = {0.f, 0.f, 0.f, 0.f, 0.f, 0.f, 0.f, 0.f};
  return c;
}
static __device__ inline v16bf v16bf_zero() {
  FragU f; u32x4 z = {0u, 0u, 0u, 0u}; f.q[0] = z; f.q[1] = z; return f.v;
}
static __device__ inline float sigmf(float x) { return 1.0f / (1.0f + __expf(-x)); }
static __device__ inline float eluf(float x)  { return x > 0.f ? x : (__expf(x) - 1.f); }

__global__ __launch_bounds__(256) void convert_f32_bf16(const float* __restrict__ s,
                                                        bf16* __restrict__ d, int n) {
  int i = blockIdx.x * 256 + threadIdx.x;
  if (i < n) d[i] = (bf16)s[i];
}

// ---------------------------------------------------------------------------
// Kernel 1: fused MLP (ELU x2) + input-side gate GEMM.
// rows = B*T (b-major). Each block: 128 rows; each wave: one 16-row M tile.
// Per-wave LDS ping-pong staging; xg written to [T,B,384] bf16 for scan locality.
// 144 WMMAs per wave (16 + 32 + 96).
// ---------------------------------------------------------------------------
__global__ __launch_bounds__(256) void k1_mlp_xg(
    const float* __restrict__ x, const bf16* __restrict__ w1, const float* __restrict__ b1,
    const bf16* __restrict__ w2, const float* __restrict__ b2,
    const bf16* __restrict__ wih, const float* __restrict__ bih,
    bf16* __restrict__ xg) {
  __shared__ bf16 bufA[8][16 * H_];
  __shared__ bf16 bufB[8][16 * H_];
  const int lane = threadIdx.x & 31;
  const int wave = threadIdx.x >> 5;
  const size_t r0 = (size_t)blockIdx.x * 128 + (size_t)wave * 16;

  // stage x tile (16x64 f32) -> bf16 LDS, pitch 64
  {
    int m = lane & 15;
    int c0 = (lane >> 4) * 32;
    const float* src = x + (r0 + m) * IN_ + c0;
    bf16* dst = &bufA[wave][m * IN_ + c0];
    for (int i = 0; i < 32; i += 8) {
      f32x4 f0 = *(const f32x4*)(src + i);
      f32x4 f1 = *(const f32x4*)(src + i + 4);
      Pack8 pk;
      #pragma unroll
      for (int j = 0; j < 4; ++j) { pk.h[j] = (bf16)f0[j]; pk.h[4 + j] = (bf16)f1[j]; }
      *(u32x4*)(dst + i) = pk.q;
    }
  }
  __syncthreads();

  // stage 1: h1 = ELU(x @ W1^T + b1)  [16x128], K=64 -> bufB
  {
    v16bf a0 = frag_a_load(&bufA[wave][0],  IN_, lane);
    v16bf a1 = frag_a_load(&bufA[wave][32], IN_, lane);
    const int n = lane & 15;
    for (int nt = 0; nt < 8; ++nt) {
      v8f c = v8f_zero();
      c = wmma_bf16(a0, frag_b_load(w1 + (size_t)(nt * 16) * IN_,      IN_, lane), c);
      c = wmma_bf16(a1, frag_b_load(w1 + (size_t)(nt * 16) * IN_ + 32, IN_, lane), c);
      float bb = b1[nt * 16 + n];
      #pragma unroll
      for (int v = 0; v < 8; ++v) {
        int m = v + ((lane >> 4) << 3);
        bufB[wave][m * H_ + nt * 16 + n] = (bf16)eluf(c[v] + bb);
      }
    }
  }
  __syncthreads();

  // stage 2: h2 = ELU(h1 @ W2^T + b2)  [16x128], K=128 -> bufA
  {
    v16bf a[4];
    #pragma unroll
    for (int k = 0; k < 4; ++k) a[k] = frag_a_load(&bufB[wave][k * 32], H_, lane);
    const int n = lane & 15;
    for (int nt = 0; nt < 8; ++nt) {
      v8f c = v8f_zero();
      #pragma unroll
      for (int k = 0; k < 4; ++k)
        c = wmma_bf16(a[k], frag_b_load(w2 + (size_t)(nt * 16) * H_ + k * 32, H_, lane), c);
      float bb = b2[nt * 16 + n];
      #pragma unroll
      for (int v = 0; v < 8; ++v) {
        int m = v + ((lane >> 4) << 3);
        bufA[wave][m * H_ + nt * 16 + n] = (bf16)eluf(c[v] + bb);
      }
    }
  }
  __syncthreads();

  // stage 3: xg = h2 @ Wih^T + bih  [16x384] -> global [T,B,384] bf16
  {
    v16bf a[4];
    #pragma unroll
    for (int k = 0; k < 4; ++k) a[k] = frag_a_load(&bufA[wave][k * 32], H_, lane);
    const int n = lane & 15;
    for (int nt = 0; nt < 24; ++nt) {
      v8f c = v8f_zero();
      #pragma unroll
      for (int k = 0; k < 4; ++k)
        c = wmma_bf16(a[k], frag_b_load(wih + (size_t)(nt * 16) * H_ + k * 32, H_, lane), c);
      float bb = bih[nt * 16 + n];
      #pragma unroll
      for (int v = 0; v < 8; ++v) {
        size_t r = r0 + v + ((lane >> 4) << 3);
        size_t t = r & (T_ - 1);   // rows are b*T + t
        size_t b = r >> 10;
        xg[(t * B_ + b) * G3_ + nt * 16 + n] = (bf16)(c[v] + bb);
      }
    }
  }
}

// ---------------------------------------------------------------------------
// Kernel 2: sequential GRU scan with per-step reset.
// 8 blocks x 256 threads; block owns 16 batch rows; hidden state resident in
// LDS (bf16 for WMMA A-frags + f32 for z*h / hT). W_hh B-fragments preloaded
// into registers (loop-invariant). Reset masks staged in LDS once. xg gate
// values loaded at iteration top so their latency hides under barrier+WMMA.
// WMMAs interleaved across the r/z/n accumulators: 3 independent chains of 4.
// ---------------------------------------------------------------------------
__global__ __launch_bounds__(256) void k2_gru(
    const bf16* __restrict__ whh, const float* __restrict__ bhh,
    const bf16* __restrict__ xg, const unsigned char* __restrict__ is_init,
    const float* __restrict__ hx, bf16* __restrict__ y, float* __restrict__ hT) {
  __shared__ bf16  hb[16 * H_];
  __shared__ float hf[16 * H_];
  __shared__ unsigned char msk[16 * T_];   // 16KB: this block's reset masks
  const int lane = threadIdx.x & 31;
  const int wave = threadIdx.x >> 5;
  const int tid  = threadIdx.x;
  const int b0   = blockIdx.x * 16;

  for (int i = tid; i < 16 * H_; i += 256) {
    float v = hx[(size_t)(b0 + (i >> 7)) * H_ + (i & 127)];
    hf[i] = v; hb[i] = (bf16)v;
  }
  {
    const u32x4* src = (const u32x4*)(is_init + (size_t)b0 * T_);
    u32x4* dst = (u32x4*)msk;
    for (int i = tid; i < (16 * T_) / 16; i += 256) dst[i] = src[i];
  }

  v16bf bw[3][4];
  float bias[3];
  #pragma unroll
  for (int g = 0; g < 3; ++g) {
    #pragma unroll
    for (int k = 0; k < 4; ++k)
      bw[g][k] = frag_b_load(whh + (size_t)((wave + 8 * g) * 16) * H_ + k * 32, H_, lane);
    bias[g] = bhh[(wave + 8 * g) * 16 + (lane & 15)];
  }
  const int colj = wave * 16 + (lane & 15);
  const int mrow = (lane >> 4) << 3;       // row base for C-fragment lanes
  __syncthreads();

  for (int t = 0; t < T_; ++t) {
    // ---- load phase (no dependence ordering on state writes of step t-1
    //      except the LDS reads, which precede the first barrier) ----
    const int mA = lane & 15;
    const bool mskA = msk[mA * T_ + t] != 0;

    // xg gate inputs for this step -> registers (global, off critical path)
    float xrv[8], xzv[8], xnv[8];
    #pragma unroll
    for (int v = 0; v < 8; ++v) {
      const bf16* xr = &xg[((size_t)t * B_ + b0 + mrow + v) * G3_ + colj];
      xrv[v] = (float)xr[0];
      xzv[v] = (float)xr[H_];
      xnv[v] = (float)xr[2 * H_];
    }
    if (t + 1 < T_)
      __builtin_prefetch(&xg[((size_t)(t + 1) * B_ + b0 + mA) * G3_], 0, 0);

    // hidden state reads (LDS)
    v16bf a[4];
    #pragma unroll
    for (int k = 0; k < 4; ++k) a[k] = frag_a_load(&hb[k * 32], H_, lane);
    if (mskA) {
      v16bf z = v16bf_zero();
      #pragma unroll
      for (int k = 0; k < 4; ++k) a[k] = z;
    }
    float heff[8];
    #pragma unroll
    for (int v = 0; v < 8; ++v) {
      int m = mrow + v;
      heff[v] = msk[m * T_ + t] ? 0.f : hf[m * H_ + colj];
    }
    __syncthreads();  // all state reads complete before overwrite

    // ---- hg = h_eff @ Whh^T : 12 WMMAs, 3 independent chains interleaved ----
    v8f cr = v8f_zero(), cz = v8f_zero(), cn = v8f_zero();
    #pragma unroll
    for (int k = 0; k < 4; ++k) {
      cr = wmma_bf16(a[k], bw[0][k], cr);
      cz = wmma_bf16(a[k], bw[1][k], cz);
      cn = wmma_bf16(a[k], bw[2][k], cn);
    }

    // ---- gates + state update (all inputs already in registers) ----
    #pragma unroll
    for (int v = 0; v < 8; ++v) {
      int m = mrow + v;
      float r  = sigmf(xrv[v] + cr[v] + bias[0]);
      float zz = sigmf(xzv[v] + cz[v] + bias[1]);
      float nn = tanhf(xnv[v] + r * (cn[v] + bias[2]));
      float hn = (1.f - zz) * nn + zz * heff[v];
      hf[m * H_ + colj] = hn;
      hb[m * H_ + colj] = (bf16)hn;
      y[((size_t)(b0 + m) * T_ + t) * H_ + colj] = (bf16)hn;
    }
    __syncthreads();  // new state visible to all waves
  }

  for (int i = tid; i < 16 * H_; i += 256)
    hT[(size_t)(b0 + (i >> 7)) * H_ + (i & 127)] = hf[i];
}

// ---------------------------------------------------------------------------
// Kernel 3: out = y @ W_out^T + b_out.  A-frags straight from global bf16 y.
// ---------------------------------------------------------------------------
__global__ __launch_bounds__(256) void k3_out(
    const bf16* __restrict__ y, const bf16* __restrict__ wout,
    const float* __restrict__ bout, float* __restrict__ out) {
  const int lane = threadIdx.x & 31;
  const int wave = threadIdx.x >> 5;
  const size_t r0 = (size_t)blockIdx.x * 128 + (size_t)wave * 16;
  v16bf a[4];
  #pragma unroll
  for (int k = 0; k < 4; ++k) a[k] = frag_a_load(y + r0 * H_ + k * 32, H_, lane);
  const int n = lane & 15;
  for (int nt = 0; nt < 4; ++nt) {
    v8f c = v8f_zero();
    #pragma unroll
    for (int k = 0; k < 4; ++k)
      c = wmma_bf16(a[k], frag_b_load(wout + (size_t)(nt * 16) * H_ + k * 32, H_, lane), c);
    float bb = bout[nt * 16 + n];
    #pragma unroll
    for (int v = 0; v < 8; ++v) {
      size_t m = r0 + v + ((lane >> 4) << 3);
      out[m * DIM_ + nt * 16 + n] = c[v] + bb;
    }
  }
}

extern "C" void kernel_launch(void* const* d_in, const int* in_sizes, int n_in,
                              void* d_out, int out_size, void* d_ws, size_t ws_size,
                              hipStream_t stream) {
  (void)in_sizes; (void)n_in; (void)out_size; (void)ws_size;
  const float* x            = (const float*)d_in[0];
  const unsigned char* isi  = (const unsigned char*)d_in[1];  // bool [B,T,1]
  const float* hx           = (const float*)d_in[2];
  const float* W1           = (const float*)d_in[3];
  const float* b1           = (const float*)d_in[4];
  const float* W2           = (const float*)d_in[5];
  const float* b2           = (const float*)d_in[6];
  const float* Wih          = (const float*)d_in[7];
  const float* bih          = (const float*)d_in[8];
  const float* Whh          = (const float*)d_in[9];
  const float* bhh          = (const float*)d_in[10];
  const float* Wout         = (const float*)d_in[11];
  const float* bout         = (const float*)d_in[12];

  // workspace layout (bf16 elements)
  bf16* w1b   = (bf16*)d_ws;
  bf16* w2b   = w1b   + 128 * 64;
  bf16* wihb  = w2b   + 128 * 128;
  bf16* whhb  = wihb  + 384 * 128;
  bf16* woutb = whhb  + 384 * 128;
  bf16* xg    = woutb + 64 * 128;                 // [T,B,384] bf16
  bf16* y     = xg + (size_t)T_ * B_ * G3_;       // [B,T,128] bf16

  float* out = (float*)d_out;                     // [B,T,64]
  float* hT  = out + (size_t)B_ * T_ * DIM_;      // [B,128]

  convert_f32_bf16<<<(128 * 64  + 255) / 256, 256, 0, stream>>>(W1,   w1b,   128 * 64);
  convert_f32_bf16<<<(128 * 128 + 255) / 256, 256, 0, stream>>>(W2,   w2b,   128 * 128);
  convert_f32_bf16<<<(384 * 128 + 255) / 256, 256, 0, stream>>>(Wih,  wihb,  384 * 128);
  convert_f32_bf16<<<(384 * 128 + 255) / 256, 256, 0, stream>>>(Whh,  whhb,  384 * 128);
  convert_f32_bf16<<<(64 * 128  + 255) / 256, 256, 0, stream>>>(Wout, woutb, 64 * 128);

  k1_mlp_xg<<<(B_ * T_) / 128, 256, 0, stream>>>(x, w1b, b1, w2b, b2, wihb, bih, xg);
  k2_gru<<<B_ / 16, 256, 0, stream>>>(whhb, bhh, xg, isi, hx, y, hT);
  k3_out<<<(B_ * T_) / 128, 256, 0, stream>>>(y, woutb, bout, out);
}
